// SparseMOE_47201690583341
// MI455X (gfx1250) — matrix operand order repaired and checked by reading, not compiled, not run on hardware
//
#include <hip/hip_runtime.h>
#include <math.h>

// ---------------------------------------------------------------------------
// Sparse MoE forward for MI455X (gfx1250, wave32, WMMA, async-to-LDS).
//   router (fp32 VALU) -> deterministic per-expert compaction
//   up:   h = gelu(x_g @ Wu_e + bu_e)    bf16 WMMA 16x16x32, f32 acc
//   down: out += g * (h_g @ Wd_e + bd_e) bf16 WMMA, f32 acc, f32 atomics
// Weights converted AND transposed once (bf16, N-major) so both GEMM operands
// stage into LDS as contiguous 16B chunks via global_load_async_to_lds_b128
// (ASYNCcnt) with double buffering; per-thread copy pointers are hoisted out
// of the K loop (padding rows -> zero page with step 0).
// Sparse routing: 137 GFLOP vs 550 GFLOP dense. Workspace ~152 MB.
// ---------------------------------------------------------------------------

typedef __attribute__((ext_vector_type(16))) __bf16 v16bf;
typedef __attribute__((ext_vector_type(8)))  __bf16 bf16x8;
typedef __attribute__((ext_vector_type(8)))  float  v8f;

constexpr int Bc = 4, Sc = 2048, Dc = 1024, Ec = 8, Hc = 2048;
constexpr int Tc = Bc * Sc;        // 8192 tokens
constexpr int Ac = Tc * 2;         // 16384 (token, k) assignments
constexpr int BM = 128, BN = 128, BK = 32;
constexpr int TM_TILES = Tc / BM;  // 64 worst-case M tiles per expert
constexpr int LDA = 40;            // padded LDS row stride (bf16): 80B, 16B-aligned
constexpr unsigned BUFB = BM * LDA * 2;  // LDS bytes per double-buffer slot

// CDNA5 async global->LDS copy (ASYNCcnt-tracked, bypasses VGPRs).
__device__ __forceinline__ void async_load_b128(unsigned lds_addr, const void* gsrc) {
  asm volatile("global_load_async_to_lds_b128 %0, %1, off"
               :: "v"(lds_addr), "v"(gsrc)
               : "memory");
}
__device__ __forceinline__ void wait_async0() {
  asm volatile("s_wait_asynccnt 0x0" ::: "memory");
}
// Low 32 bits of a generic pointer to __shared__ are the LDS byte address.
__device__ __forceinline__ unsigned lds_addr32(const void* p) {
  return (unsigned)(unsigned long long)p;
}

// --------------------------- fp32 -> bf16 conversion -----------------------
__global__ void moe_cvt_bf16(const float* __restrict__ src, __bf16* __restrict__ dst, int n) {
  int stride = gridDim.x * blockDim.x;
  for (int i = blockIdx.x * blockDim.x + threadIdx.x; i < n; i += stride)
    dst[i] = (__bf16)src[i];
}

// ----------------- fp32 -> bf16 with transpose (per expert) ----------------
// src: [E][R][C] fp32, dst: [E][C][R] bf16. grid = (C/64, R/64, E).
__global__ void __launch_bounds__(256)
moe_cvt_t(const float* __restrict__ src, __bf16* __restrict__ dst, int R, int C) {
  __shared__ __bf16 tile[64][65];
  const int e = blockIdx.z;
  const int c0 = blockIdx.x * 64, r0 = blockIdx.y * 64;
  const float* s = src + (size_t)e * R * C;
  __bf16* d = dst + (size_t)e * R * C;
#pragma unroll
  for (int j = 0; j < 16; ++j) {
    int idx = threadIdx.x + j * 256;
    int r = idx >> 6, c = idx & 63;
    tile[r][c] = (__bf16)s[(size_t)(r0 + r) * C + c0 + c];
  }
  __syncthreads();
#pragma unroll
  for (int j = 0; j < 16; ++j) {
    int idx = threadIdx.x + j * 256;
    int c = idx >> 6, r = idx & 63;
    d[(size_t)(c0 + c) * R + r0 + r] = tile[r][c];
  }
}

// --------------------------- noisy top-2 router ----------------------------
__global__ void __launch_bounds__(256)
moe_router(const float* __restrict__ x, const float* __restrict__ noise,
           const float* __restrict__ Wg, const float* __restrict__ bg,
           const float* __restrict__ Wn, const float* __restrict__ bn,
           int* __restrict__ sel0, int* __restrict__ sel1, float* __restrict__ gates) {
  const int lane = threadIdx.x & 31;
  const int wid  = threadIdx.x >> 5;
  const int t = blockIdx.x * (blockDim.x >> 5) + wid;
  if (t >= Tc) return;

  float ag[Ec], an[Ec];
#pragma unroll
  for (int e = 0; e < Ec; ++e) { ag[e] = 0.f; an[e] = 0.f; }

  const float* xr = x + (size_t)t * Dc;
  for (int d = lane; d < Dc; d += 32) {
    float xv = xr[d];
    const float* wg = Wg + d * Ec;
    const float* wn = Wn + d * Ec;
#pragma unroll
    for (int e = 0; e < Ec; ++e) { ag[e] += xv * wg[e]; an[e] += xv * wn[e]; }
  }
#pragma unroll
  for (int e = 0; e < Ec; ++e) {
#pragma unroll
    for (int off = 16; off > 0; off >>= 1) {
      ag[e] += __shfl_xor(ag[e], off, 32);
      an[e] += __shfl_xor(an[e], off, 32);
    }
  }
  float nv[Ec];
#pragma unroll
  for (int e = 0; e < Ec; ++e) {
    float lg = ag[e] + bg[e];
    float nl = an[e] + bn[e];
    float sp = (nl > 20.f) ? nl : log1pf(expf(nl));
    nv[e] = lg + noise[(size_t)t * Ec + e] * sp;
  }
  int e0 = 0;
#pragma unroll
  for (int e = 1; e < Ec; ++e) if (nv[e] > nv[e0]) e0 = e;
  int e1 = (e0 == 0) ? 1 : 0;
#pragma unroll
  for (int e = 0; e < Ec; ++e) if (e != e0 && nv[e] > nv[e1]) e1 = e;
  float ex = expf(nv[e1] - nv[e0]);
  float g0 = 1.f / (1.f + ex);
  float g1 = ex / (1.f + ex);
  if (lane == 0) {
    sel0[t] = e0; sel1[t] = e1;
    gates[t * 2 + 0] = g0; gates[t * 2 + 1] = g1;
  }
}

// ----------------- deterministic per-expert compaction ---------------------
__global__ void __launch_bounds__(256)
moe_compact(const int* __restrict__ sel0, const int* __restrict__ sel1,
            int* __restrict__ lists, int* __restrict__ counts) {
  const int e = blockIdx.x;
  const int lane = threadIdx.x & 31, wid = threadIdx.x >> 5;
  __shared__ int base;
  __shared__ int wsum[8];
  if (threadIdx.x == 0) base = 0;
  __syncthreads();
  for (int start = 0; start < Tc; start += 256) {
    int t = start + threadIdx.x;
    bool f0 = (sel0[t] == e);
    bool f1 = (sel1[t] == e);
    bool f  = f0 || f1;
    unsigned mask = (unsigned)__ballot(f);
    int rank = __popc(mask & ((1u << lane) - 1u));
    if (lane == 0) wsum[wid] = __popc(mask);
    __syncthreads();
    int woff = 0;
    for (int w = 0; w < wid; ++w) woff += wsum[w];
    if (f) lists[e * Tc + base + woff + rank] = t * 2 + (f0 ? 0 : 1);
    __syncthreads();
    if (threadIdx.x == 0) {
      int tot = 0;
      for (int w = 0; w < 8; ++w) tot += wsum[w];
      base += tot;
    }
    __syncthreads();
  }
  if (threadIdx.x == 0) counts[e] = base;
}

struct FragSet {
  union V16 { v16bf v; bf16x8 h[2]; };
  V16 a[2], b[4];
};

// --------------------------- up projection GEMM ----------------------------
__global__ void __launch_bounds__(256)
moe_up(const __bf16* __restrict__ xb, const __bf16* __restrict__ Wut,
       const float* __restrict__ bu, const int* __restrict__ lists,
       const int* __restrict__ counts, const __bf16* __restrict__ zpad,
       __bf16* __restrict__ hb) {
  const int e  = blockIdx.x / TM_TILES;
  const int tm = blockIdx.x % TM_TILES;
  const int cnt = counts[e];
  if (tm * BM >= cnt) return;
  const int n0 = blockIdx.y * BN;

  __shared__ __align__(16) __bf16 As[2][BM][LDA];
  __shared__ __align__(16) __bf16 Bs[2][BN][LDA];
  __shared__ int rowid[BM];

  const int tid = threadIdx.x;
  const int lane = tid & 31, wid = tid >> 5;
  const int waveN = wid & 1, waveM = wid >> 1;
  const int l15 = lane & 15;
  const int hiA = (lane < 16) ? 0 : 8;
  const int hiB = (lane < 16) ? 0 : 8;
  const int mhi = (lane < 16) ? 0 : 8;

  if (tid < BM) {
    int g = tm * BM + tid;
    rowid[tid] = (g < cnt) ? lists[e * Tc + g] : -1;
  }
  __syncthreads();

  const __bf16* Wt = Wut + (size_t)e * Hc * Dc;   // [H][D]

  // ---- loop-invariant per-thread staging descriptors ----
  const __bf16* aPtr[2];
  const __bf16* bPtr[2];
  int aStep[2];
  unsigned aL0[2], bL0[2];
#pragma unroll
  for (int j = 0; j < 2; ++j) {
    int c = tid + j * 256;               // 0..511: 4 chunks of 8 bf16 per row
    int r = c >> 2, kc = (c & 3) * 8;
    int rv = rowid[r];
    aPtr[j]  = (rv >= 0) ? (xb + (size_t)(rv >> 1) * Dc + kc) : (zpad + kc);
    aStep[j] = (rv >= 0) ? BK : 0;       // padding rows never advance
    bPtr[j]  = Wt + (size_t)(n0 + r) * Dc + kc;
    aL0[j] = lds_addr32(&As[0][r][kc]);
    bL0[j] = lds_addr32(&Bs[0][r][kc]);
  }

  auto stage = [&](int buf) {
    unsigned boff = (unsigned)buf * BUFB;
#pragma unroll
    for (int j = 0; j < 2; ++j) {
      async_load_b128(aL0[j] + boff, aPtr[j]);
      async_load_b128(bL0[j] + boff, bPtr[j]);
      aPtr[j] += aStep[j];
      bPtr[j] += BK;
    }
  };

  v8f acc[2][4];
#pragma unroll
  for (int mt = 0; mt < 2; ++mt)
#pragma unroll
    for (int nt = 0; nt < 4; ++nt)
#pragma unroll
      for (int q = 0; q < 8; ++q) acc[mt][nt][q] = 0.f;

  stage(0);
  wait_async0();
  __syncthreads();

  int cur = 0;
  for (int kk = 0; kk < Dc; kk += BK) {
    if (kk + BK < Dc) stage(cur ^ 1);    // overlap next-tile copy with math

    FragSet f;
#pragma unroll
    for (int mt = 0; mt < 2; ++mt) {
      int r = waveM * 32 + mt * 16 + l15;
      f.a[mt].h[0] = *(const bf16x8*)&As[cur][r][hiA];
      f.a[mt].h[1] = *(const bf16x8*)&As[cur][r][hiA + 16];
    }
#pragma unroll
    for (int nt = 0; nt < 4; ++nt) {
      int ncol = waveN * 64 + nt * 16 + l15;
      f.b[nt].h[0] = *(const bf16x8*)&Bs[cur][ncol][hiB];
      f.b[nt].h[1] = *(const bf16x8*)&Bs[cur][ncol][hiB + 16];
    }
#pragma unroll
    for (int mt = 0; mt < 2; ++mt)
#pragma unroll
      for (int nt = 0; nt < 4; ++nt)
        acc[mt][nt] = __builtin_amdgcn_wmma_f32_16x16x32_bf16(
            false, f.a[mt].v, false, f.b[nt].v, (short)0, acc[mt][nt], false, false);

    wait_async0();
    __syncthreads();
    cur ^= 1;
  }

  // epilogue: + bu, exact GELU, bf16 store to h rows
#pragma unroll
  for (int mt = 0; mt < 2; ++mt) {
#pragma unroll
    for (int nt = 0; nt < 4; ++nt) {
      int ncol = n0 + waveN * 64 + nt * 16 + l15;
      float bias = bu[e * Hc + ncol];
#pragma unroll
      for (int i = 0; i < 8; ++i) {
        int m = waveM * 32 + mt * 16 + mhi + i;
        int rv = rowid[m];
        if (rv >= 0) {
          float v = acc[mt][nt][i] + bias;
          v = 0.5f * v * (1.0f + erff(v * 0.70710678118654752f));
          hb[(size_t)rv * Hc + ncol] = (__bf16)v;
        }
      }
    }
  }
}

// --------------------------- down projection GEMM --------------------------
__global__ void __launch_bounds__(256)
moe_down(const __bf16* __restrict__ hb, const __bf16* __restrict__ Wdt,
         const float* __restrict__ bd, const int* __restrict__ lists,
         const int* __restrict__ counts, const float* __restrict__ gates,
         const __bf16* __restrict__ zpad, float* __restrict__ out) {
  const int e  = blockIdx.x / TM_TILES;
  const int tm = blockIdx.x % TM_TILES;
  const int cnt = counts[e];
  if (tm * BM >= cnt) return;
  const int n0 = blockIdx.y * BN;

  __shared__ __align__(16) __bf16 As[2][BM][LDA];
  __shared__ __align__(16) __bf16 Bs[2][BN][LDA];
  __shared__ int rowid[BM];

  const int tid = threadIdx.x;
  const int lane = tid & 31, wid = tid >> 5;
  const int waveN = wid & 1, waveM = wid >> 1;
  const int l15 = lane & 15;
  const int hiA = (lane < 16) ? 0 : 8;
  const int hiB = (lane < 16) ? 0 : 8;
  const int mhi = (lane < 16) ? 0 : 8;

  if (tid < BM) {
    int g = tm * BM + tid;
    rowid[tid] = (g < cnt) ? lists[e * Tc + g] : -1;
  }
  __syncthreads();

  const __bf16* Wt = Wdt + (size_t)e * Dc * Hc;   // [D][H]

  const __bf16* aPtr[2];
  const __bf16* bPtr[2];
  int aStep[2];
  unsigned aL0[2], bL0[2];
#pragma unroll
  for (int j = 0; j < 2; ++j) {
    int c = tid + j * 256;
    int r = c >> 2, kc = (c & 3) * 8;
    int rv = rowid[r];
    aPtr[j]  = (rv >= 0) ? (hb + (size_t)rv * Hc + kc) : (zpad + kc);
    aStep[j] = (rv >= 0) ? BK : 0;
    bPtr[j]  = Wt + (size_t)(n0 + r) * Hc + kc;
    aL0[j] = lds_addr32(&As[0][r][kc]);
    bL0[j] = lds_addr32(&Bs[0][r][kc]);
  }

  auto stage = [&](int buf) {
    unsigned boff = (unsigned)buf * BUFB;
#pragma unroll
    for (int j = 0; j < 2; ++j) {
      async_load_b128(aL0[j] + boff, aPtr[j]);
      async_load_b128(bL0[j] + boff, bPtr[j]);
      aPtr[j] += aStep[j];
      bPtr[j] += BK;
    }
  };

  v8f acc[2][4];
#pragma unroll
  for (int mt = 0; mt < 2; ++mt)
#pragma unroll
    for (int nt = 0; nt < 4; ++nt)
#pragma unroll
      for (int q = 0; q < 8; ++q) acc[mt][nt][q] = 0.f;

  stage(0);
  wait_async0();
  __syncthreads();

  int cur = 0;
  for (int kk = 0; kk < Hc; kk += BK) {
    if (kk + BK < Hc) stage(cur ^ 1);

    FragSet f;
#pragma unroll
    for (int mt = 0; mt < 2; ++mt) {
      int r = waveM * 32 + mt * 16 + l15;
      f.a[mt].h[0] = *(const bf16x8*)&As[cur][r][hiA];
      f.a[mt].h[1] = *(const bf16x8*)&As[cur][r][hiA + 16];
    }
#pragma unroll
    for (int nt = 0; nt < 4; ++nt) {
      int ncol = waveN * 64 + nt * 16 + l15;
      f.b[nt].h[0] = *(const bf16x8*)&Bs[cur][ncol][hiB];
      f.b[nt].h[1] = *(const bf16x8*)&Bs[cur][ncol][hiB + 16];
    }
#pragma unroll
    for (int mt = 0; mt < 2; ++mt)
#pragma unroll
      for (int nt = 0; nt < 4; ++nt)
        acc[mt][nt] = __builtin_amdgcn_wmma_f32_16x16x32_bf16(
            false, f.a[mt].v, false, f.b[nt].v, (short)0, acc[mt][nt], false, false);

    wait_async0();
    __syncthreads();
    cur ^= 1;
  }

#pragma unroll
  for (int mt = 0; mt < 2; ++mt) {
#pragma unroll
    for (int nt = 0; nt < 4; ++nt) {
      int ncol = n0 + waveN * 64 + nt * 16 + l15;
      float bias = bd[e * Dc + ncol];
#pragma unroll
      for (int i = 0; i < 8; ++i) {
        int m = waveM * 32 + mt * 16 + mhi + i;
        int rv = rowid[m];
        if (rv >= 0) {
          float v = acc[mt][nt][i] + bias;
          atomicAdd(out + (size_t)(rv >> 1) * Dc + ncol, gates[rv] * v);
        }
      }
    }
  }
}

// ------------------------------- launcher ----------------------------------
extern "C" void kernel_launch(void* const* d_in, const int* in_sizes, int n_in,
                              void* d_out, int out_size, void* d_ws, size_t ws_size,
                              hipStream_t stream) {
  const float* x     = (const float*)d_in[0];
  const float* noise = (const float*)d_in[1];
  const float* Wg    = (const float*)d_in[2];
  const float* bg    = (const float*)d_in[3];
  const float* Wn    = (const float*)d_in[4];
  const float* bn    = (const float*)d_in[5];
  const float* Wu    = (const float*)d_in[6];
  const float* bu    = (const float*)d_in[7];
  const float* Wd    = (const float*)d_in[8];
  const float* bd    = (const float*)d_in[9];
  float* out = (float*)d_out;
  (void)in_sizes; (void)n_in; (void)out_size; (void)ws_size;

  char* ws = (char*)d_ws;
  size_t off = 0;
  auto take = [&](size_t bytes) {
    char* p = ws + off;
    off += (bytes + 255) & ~(size_t)255;
    return p;
  };
  __bf16* xb    = (__bf16*)take((size_t)Tc * Dc * 2);        // 16.8 MB
  __bf16* Wut   = (__bf16*)take((size_t)Ec * Dc * Hc * 2);   // 33.6 MB [E][H][D]
  __bf16* Wdt   = (__bf16*)take((size_t)Ec * Hc * Dc * 2);   // 33.6 MB [E][D][H]
  __bf16* hb    = (__bf16*)take((size_t)Ac * Hc * 2);        // 67.1 MB
  int*    sel0  = (int*)take((size_t)Tc * 4);
  int*    sel1  = (int*)take((size_t)Tc * 4);
  float*  gates = (float*)take((size_t)Ac * 4);
  int*    lists = (int*)take((size_t)Ec * Tc * 4);
  int*    counts= (int*)take((size_t)Ec * 4);
  __bf16* zpad  = (__bf16*)take(256);                         // zero page for pad rows

  hipMemsetAsync(d_out, 0, (size_t)Tc * Dc * sizeof(float), stream);
  hipMemsetAsync(zpad, 0, 256, stream);

  moe_cvt_bf16<<<2048, 256, 0, stream>>>(x, xb, Tc * Dc);
  // Wu [E][D][H] -> Wut [E][H][D]; Wd [E][H][D] -> Wdt [E][D][H]
  moe_cvt_t<<<dim3(Hc / 64, Dc / 64, Ec), 256, 0, stream>>>(Wu, Wut, Dc, Hc);
  moe_cvt_t<<<dim3(Dc / 64, Hc / 64, Ec), 256, 0, stream>>>(Wd, Wdt, Hc, Dc);

  moe_router<<<Tc / 8, 256, 0, stream>>>(x, noise, Wg, bg, Wn, bn, sel0, sel1, gates);
  moe_compact<<<Ec, 256, 0, stream>>>(sel0, sel1, lists, counts);

  moe_up<<<dim3(Ec * TM_TILES, Hc / BN), 256, 0, stream>>>(xb, Wut, bu, lists, counts, zpad, hb);
  moe_down<<<dim3(Ec * TM_TILES, Dc / BN), 256, 0, stream>>>(hb, Wdt, bd, lists, counts, gates, zpad, out);
}
